// ToRGB_28346784153761
// MI455X (gfx1250) — matrix operand order repaired and checked by reading, not compiled, not run on hardware
//
#include <hip/hip_runtime.h>
#include <hip/hip_bf16.h>
#include <math.h>

typedef __attribute__((ext_vector_type(16))) _Float16 v16h;
typedef __attribute__((ext_vector_type(8)))  float    v8f;
typedef __attribute__((ext_vector_type(4)))  float    f32x4;

#define BATCH 8
#define HW    16384      // 128*128 pixels per batch image
#define CIN   512
#define COUT  4
#define WDIM  128
#define NPAD  16         // N padded from 4 -> 16 for WMMA
#define KBLK  32
#define NKBLK (CIN / KBLK)   // 16 K-blocks of 32

// Workspace: B fragments in exact WMMA B-operand layout, f16:
//   [BATCH][NKBLK][32 lanes][16 halves]  = 8*16*32*16 halves = 128 KB

// ---------------------------------------------------------------------------
// Kernel 1: style affine + fold scale into 1x1 kernel, scatter into the
// v_wmma_f32_16x16x32_f16 B-matrix (32x16, K-major per lane-half) layout.
// B[k][n]: lane = ((k>>3)&1)*16 + n ; element e = (k & 7) | ((k>>4)<<3).
// ---------------------------------------------------------------------------
__global__ void torgb_precompute(const float* __restrict__ w,
                                 const float* __restrict__ affine_w,
                                 const float* __restrict__ affine_b,
                                 const float* __restrict__ kernel,
                                 _Float16* __restrict__ wsB)
{
    const int b = blockIdx.x;      // batch
    const int c = threadIdx.x;     // input channel 0..511

    // s[b][c] = (w[b,:] . affine_w[:,c] + affine_b[c]) / sqrt(512)
    float s = affine_b[c];
    const float* wb = w + b * WDIM;
    #pragma unroll 4
    for (int k = 0; k < WDIM; ++k)
        s = fmaf(wb[k], affine_w[k * CIN + c], s);
    s *= 0.044194173824159216f;    // 1/sqrt(512)

    const int kk = c >> 5;         // K-block
    const int kr = c & 31;         // k within block
    const int h  = (kr >> 3) & 1;  // which lane-half holds this k row of B
    const int e  = (kr & 7) | ((kr >> 4) << 3); // element index inside v16h

    _Float16* dst = wsB + ((size_t)(b * NKBLK + kk) * 32) * 16;
    #pragma unroll
    for (int n = 0; n < NPAD; ++n) {
        float v = (n < COUT) ? s * kernel[c * COUT + n] : 0.0f;
        dst[(h * 16 + n) * 16 + e] = (_Float16)v;
    }
}

// ---------------------------------------------------------------------------
// Kernel 2: streaming modulated 1x1 conv + tanh.
// One wave32 per 16-pixel tile; K-loop of 16 WMMAs (16x16x32 f16 -> f32).
// A-fragment layout (16-bit A 16x32): lane half h holds K runs
// [h*8 .. h*8+7] and [16+h*8 .. 16+h*8+7]  -> two contiguous 32B reads;
// lanes L and L+16 jointly cover one full 128B cacheline per row per K-block.
// ---------------------------------------------------------------------------
__global__ void __launch_bounds__(256)
torgb_main(const float* __restrict__ x,
           const _Float16* __restrict__ wsB,
           const float* __restrict__ bias,
           float* __restrict__ out)
{
    const int lane = threadIdx.x & 31;
    const int wave = threadIdx.x >> 5;
    const int tile = blockIdx.x * 8 + wave;   // 0..8191
    const int b    = tile >> 10;              // tile / 1024
    const int t    = tile & 1023;
    const int pix0 = t * 16;
    const int M    = lane & 15;               // A row (pixel within tile)
    const int h    = lane >> 4;               // lane half
    const int N    = lane & 15;               // C/D column

    const float*    xrow = x + ((size_t)b * HW + pix0 + M) * CIN;
    const _Float16* bfp  = wsB + ((size_t)b * NKBLK * 32 + lane) * 16;

    // Fold bias into accumulator init (all 8 C regs of a lane share column N).
    float biasv = bias[N & 3];
    if (N >= COUT) biasv = 0.0f;
    v8f acc;
    #pragma unroll
    for (int i = 0; i < 8; ++i) acc[i] = biasv;

    #pragma unroll
    for (int kk = 0; kk < NKBLK; ++kk) {
        const f32x4* p0 = (const f32x4*)(xrow + kk * KBLK + h * 8);
        const f32x4* p1 = (const f32x4*)(xrow + kk * KBLK + 16 + h * 8);
        // x is single-use (256MB stream): bypass cache retention.
        f32x4 r0 = __builtin_nontemporal_load(p0);
        f32x4 r1 = __builtin_nontemporal_load(p0 + 1);
        f32x4 r2 = __builtin_nontemporal_load(p1);
        f32x4 r3 = __builtin_nontemporal_load(p1 + 1);

        v16h a;
        a[0]  = (_Float16)r0.x;  a[1]  = (_Float16)r0.y;
        a[2]  = (_Float16)r0.z;  a[3]  = (_Float16)r0.w;
        a[4]  = (_Float16)r1.x;  a[5]  = (_Float16)r1.y;
        a[6]  = (_Float16)r1.z;  a[7]  = (_Float16)r1.w;
        a[8]  = (_Float16)r2.x;  a[9]  = (_Float16)r2.y;
        a[10] = (_Float16)r2.z;  a[11] = (_Float16)r2.w;
        a[12] = (_Float16)r3.x;  a[13] = (_Float16)r3.y;
        a[14] = (_Float16)r3.z;  a[15] = (_Float16)r3.w;

        v16h bf = *(const v16h*)(bfp + (size_t)kk * 32 * 16);

        acc = __builtin_amdgcn_wmma_f32_16x16x32_f16(
            /*neg_a=*/false, a, /*neg_b=*/false, bf,
            /*c_mod=*/(short)0, acc, /*reuse_a=*/false, /*reuse_b=*/false);
    }

    if (N < COUT) {
        #pragma unroll
        for (int r = 0; r < 8; ++r) {
            const int Mr = r + 8 * h;           // C/D row layout
            out[((size_t)b * HW + pix0 + Mr) * COUT + N] = tanhf(acc[r]);
        }
    }
}

extern "C" void kernel_launch(void* const* d_in, const int* in_sizes, int n_in,
                              void* d_out, int out_size, void* d_ws, size_t ws_size,
                              hipStream_t stream) {
    const float* x      = (const float*)d_in[0];
    const float* w      = (const float*)d_in[1];
    const float* aw     = (const float*)d_in[2];
    const float* ab     = (const float*)d_in[3];
    const float* kern   = (const float*)d_in[4];
    const float* bias   = (const float*)d_in[5];
    float*       out    = (float*)d_out;
    _Float16*    wsB    = (_Float16*)d_ws;     // 128 KB used

    torgb_precompute<<<BATCH, CIN, 0, stream>>>(w, aw, ab, kern, wsB);
    torgb_main<<<1024, 256, 0, stream>>>(x, wsB, bias, out);
}
